// OptimizedS6SSMEncoder_30571577213791
// MI455X (gfx1250) — matrix operand order repaired and checked by reading, not compiled, not run on hardware
//
#include <hip/hip_runtime.h>
#include <hip/hip_bf16.h>
#include <math.h>

// ---------------------------------------------------------------------------
// S6 / Mamba encoder for MI455X (gfx1250, wave32).
// GEMMs: v_wmma_f32_16x16x32_bf16, 2x2 register tiling (32x32 out per wave):
//        4 WMMAs per 8 global_load_b128 -> 2x the FLOP/byte of 1-tile waves.
// Scan : one wave per (b, channel), 2 states/lane, shfl_xor butterfly.
// ---------------------------------------------------------------------------

#define D_MODEL 768
#define D_STATE 64
#define D_INNER 1536          // 2*D_MODEL
#define XDBL    1664          // 2*D_STATE + D_INNER
#define N_LAYERS 3
#define BATCH   2
#define SEQ     512
#define M_TOK   (BATCH*SEQ)   // 1024 rows

typedef __bf16 bf16_t;
typedef __attribute__((ext_vector_type(16))) __bf16 v16bf;
typedef __attribute__((ext_vector_type(8)))  float  v8f;

union FragAB { v16bf v; uint4 q[2]; };

// ---------------------------------------------------------------- converts
__global__ void k_cvt_bf16(const float* __restrict__ s, bf16_t* __restrict__ d, int n) {
    int i = blockIdx.x * blockDim.x + threadIdx.x;
    if (i < n) d[i] = (bf16_t)s[i];
}

// ---------------------------------------------------------------- WMMA GEMM
// C[M,N] = A[M,K] * W[N,K]^T   (A, W bf16 row-major; C f32 row-major)
// M%32==0, N%128==0, K%32==0. Block = 128 thr = 4 waves; wave = 32x32 tile.
__global__ void k_gemm_wmma_bf16(const bf16_t* __restrict__ A,
                                 const bf16_t* __restrict__ W,
                                 float* __restrict__ C,
                                 int M, int N, int K) {
    const int lane = threadIdx.x & 31;
    const int wave = threadIdx.x >> 5;
    const int m0   = blockIdx.y * 32;                  // wave's 32-row strip
    const int n0   = (blockIdx.x * 4 + wave) * 32;     // wave's 32-col strip
    if (m0 >= M || n0 >= N) return;                    // wave-uniform

    const bool hi  = lane >= 16;
    const int  l15 = lane & 15;
    const int  ka_off = hi ? 8  : 0;   // ISA A-layout (16-bit, 16x32)
    const int  kb_off = hi ? 8  : 0;   // ISA B-layout (16-bit, 32x16)

    const bf16_t* __restrict__ arow0 = A + (size_t)(m0 + l15) * K + ka_off;
    const bf16_t* __restrict__ arow1 = arow0 + (size_t)16 * K;
    const bf16_t* __restrict__ brow0 = W + (size_t)(n0 + l15) * K + kb_off;
    const bf16_t* __restrict__ brow1 = brow0 + (size_t)16 * K;

    v8f acc00 = {}, acc01 = {}, acc10 = {}, acc11 = {};

    for (int k0 = 0; k0 < K; k0 += 32) {
        FragAB a0, a1, b0, b1;
        a0.q[0] = *(const uint4*)(arow0 + k0);         // K = base .. base+7
        a0.q[1] = *(const uint4*)(arow0 + k0 + 16);    // K = base+16 .. +23
        a1.q[0] = *(const uint4*)(arow1 + k0);
        a1.q[1] = *(const uint4*)(arow1 + k0 + 16);
        b0.q[0] = *(const uint4*)(brow0 + k0);         // K = base .. base+7
        b0.q[1] = *(const uint4*)(brow0 + k0 + 16);    // K = base+16 .. +23
        b1.q[0] = *(const uint4*)(brow1 + k0);
        b1.q[1] = *(const uint4*)(brow1 + k0 + 16);

        acc00 = __builtin_amdgcn_wmma_f32_16x16x32_bf16(
                    false, a0.v, false, b0.v, (short)0, acc00, false, false);
        acc01 = __builtin_amdgcn_wmma_f32_16x16x32_bf16(
                    false, a0.v, false, b1.v, (short)0, acc01, false, false);
        acc10 = __builtin_amdgcn_wmma_f32_16x16x32_bf16(
                    false, a1.v, false, b0.v, (short)0, acc10, false, false);
        acc11 = __builtin_amdgcn_wmma_f32_16x16x32_bf16(
                    false, a1.v, false, b1.v, (short)0, acc11, false, false);
    }

    // D layout: VGPR r -> row r + (hi?8:0); col = lane&15
    const int mb  = m0 + (hi ? 8 : 0);
    const int nc0 = n0 + l15;
#pragma unroll
    for (int r = 0; r < 8; ++r) {
        C[(size_t)(mb + r)      * N + nc0     ] = acc00[r];
        C[(size_t)(mb + r)      * N + nc0 + 16] = acc01[r];
        C[(size_t)(mb + r + 16) * N + nc0     ] = acc10[r];
        C[(size_t)(mb + r + 16) * N + nc0 + 16] = acc11[r];
    }
}

// ---------------------------------------------------------------- elementwise
__device__ __forceinline__ float silu_f(float v) {
    return v / (1.0f + __expf(-v));
}

// xr[M,3072] -> xi_f32[M,1536] = silu(first half), xi_bf16 (for x_proj GEMM)
__global__ void k_silu_split(const float* __restrict__ xr,
                             float* __restrict__ xi_f,
                             bf16_t* __restrict__ xi_bf, int n) {
    int i = blockIdx.x * blockDim.x + threadIdx.x;
    if (i >= n) return;
    int row = i / D_INNER, col = i - row * D_INNER;
    float v = silu_f(xr[(size_t)row * (2 * D_INNER) + col]);
    xi_f[i]  = v;
    xi_bf[i] = (bf16_t)v;
}

// x_dbl[M,1664] cols 64..127 -> bf16 [M,64]
__global__ void k_extract_dltin(const float* __restrict__ xdbl,
                                bf16_t* __restrict__ dlt, int n) {
    int i = blockIdx.x * blockDim.x + threadIdx.x;
    if (i >= n) return;
    int row = i >> 6, col = i & 63;
    dlt[i] = (bf16_t)xdbl[(size_t)row * XDBL + D_STATE + col];
}

// delta = min(softplus(dt_raw + dt_b), 10)   in place
__global__ void k_delta(float* __restrict__ dt_raw,
                        const float* __restrict__ dt_b, int n) {
    int i = blockIdx.x * blockDim.x + threadIdx.x;
    if (i >= n) return;
    int col = i % D_INNER;
    float v  = dt_raw[i] + dt_b[col];
    float sp = (v > 20.0f) ? v : log1pf(__expf(v));
    dt_raw[i] = fminf(sp, 10.0f);
}

// y = (y_scan + xi*Dp) * silu(res);   emit bf16 for out_proj GEMM
__global__ void k_ypost(const float* __restrict__ y_scan,
                        const float* __restrict__ xi,
                        const float* __restrict__ xr,      // res = second half
                        const float* __restrict__ Dp,
                        bf16_t* __restrict__ y_bf, int n) {
    int i = blockIdx.x * blockDim.x + threadIdx.x;
    if (i >= n) return;
    int row = i / D_INNER, col = i - row * D_INNER;
    float ys = y_scan[i] + xi[i] * Dp[col];
    float r  = xr[(size_t)row * (2 * D_INNER) + D_INNER + col];
    y_bf[i]  = (bf16_t)(ys * silu_f(r));
}

// ---------------------------------------------------------------- scan
// One wave per (b, e).  Lane holds states s = 2*lane, 2*lane+1.
__global__ void k_scan(const float* __restrict__ delta,   // [B*L, 1536]
                       const float* __restrict__ xdbl,    // [B*L, 1664]
                       const float* __restrict__ xi,      // [B*L, 1536]
                       const float* __restrict__ A_log,   // [1536, 64] (layer)
                       float* __restrict__ y) {           // [B*L, 1536]
    const int lane = threadIdx.x & 31;
    const int g    = blockIdx.x * (blockDim.x >> 5) + (threadIdx.x >> 5);
    if (g >= BATCH * D_INNER) return;
    const int b = g / D_INNER;
    const int e = g - b * D_INNER;

    float2 al = *(const float2*)&A_log[(size_t)e * D_STATE + lane * 2];
    const float A0 = -__expf(al.x);
    const float A1 = -__expf(al.y);

    float h0 = 0.0f, h1 = 0.0f;
    for (int t = 0; t < SEQ; ++t) {
        const size_t row = (size_t)b * SEQ + t;
        const float d  = delta[row * D_INNER + e];
        const float xv = xi[row * D_INNER + e];
        const float Cv = xdbl[row * XDBL + 2 * D_STATE + e];
        float2 Bv = *(const float2*)&xdbl[row * XDBL + lane * 2];

        const float dBx = d * xv;
        h0 = __expf(d * A0) * h0 + dBx * Bv.x;
        h1 = __expf(d * A1) * h1 + dBx * Bv.y;

        float part = h0 + h1;
#pragma unroll
        for (int m = 16; m >= 1; m >>= 1)
            part += __shfl_xor(part, m, 32);

        if (lane == 0) y[row * D_INNER + e] = Cv * part;
    }
}

// ---------------------------------------------------------------- layernorm
// One block (256 thr) per row of 768.  dst = LN(src [+ resid]) * g + b
__global__ void k_add_ln(const float* __restrict__ src,
                         const float* __restrict__ resid,   // may be null
                         const float* __restrict__ g,
                         const float* __restrict__ b,
                         float* __restrict__ dst) {
    __shared__ float red[256];
    const int row = blockIdx.x, t = threadIdx.x;
    const size_t base = (size_t)row * D_MODEL;

    float v[3]; float s = 0.0f;
#pragma unroll
    for (int j = 0; j < 3; ++j) {
        int c = t + j * 256;
        float x = src[base + c];
        if (resid) x += resid[base + c];
        v[j] = x; s += x;
    }
    red[t] = s; __syncthreads();
    for (int off = 128; off > 0; off >>= 1) {
        if (t < off) red[t] += red[t + off];
        __syncthreads();
    }
    const float mu = red[0] * (1.0f / D_MODEL);
    __syncthreads();

    float vs = 0.0f;
#pragma unroll
    for (int j = 0; j < 3; ++j) { float d = v[j] - mu; vs += d * d; }
    red[t] = vs; __syncthreads();
    for (int off = 128; off > 0; off >>= 1) {
        if (t < off) red[t] += red[t + off];
        __syncthreads();
    }
    const float rs = rsqrtf(red[0] * (1.0f / D_MODEL) + 1e-5f);

#pragma unroll
    for (int j = 0; j < 3; ++j) {
        int c = t + j * 256;
        dst[base + c] = (v[j] - mu) * rs * g[c] + b[c];
    }
}

// ---------------------------------------------------------------- launch
static inline size_t alignUp(size_t x) { return (x + 255) & ~(size_t)255; }

extern "C" void kernel_launch(void* const* d_in, const int* in_sizes, int n_in,
                              void* d_out, int out_size, void* d_ws, size_t ws_size,
                              hipStream_t stream) {
    (void)in_sizes; (void)n_in; (void)out_size; (void)ws_size;

    const float* x_in   = (const float*)d_in[0];
    const float* in_w   = (const float*)d_in[1];   // [3, 3072, 768]
    const float* xp_w   = (const float*)d_in[2];   // [3, 1664, 1536]
    const float* dt_w   = (const float*)d_in[3];   // [3, 1536, 64]
    const float* dt_b   = (const float*)d_in[4];   // [3, 1536]
    const float* A_log  = (const float*)d_in[5];   // [3, 1536, 64]
    const float* D_par  = (const float*)d_in[6];   // [3, 1536]
    const float* out_w  = (const float*)d_in[7];   // [3, 768, 1536]
    const float* ln_g   = (const float*)d_in[8];   // [3, 768]
    const float* ln_b   = (const float*)d_in[9];
    const float* fln_g  = (const float*)d_in[10];
    const float* fln_b  = (const float*)d_in[11];
    float* out = (float*)d_out;

    // ----- workspace carve (bump allocator, 256B aligned), ~65 MB total
    char* p = (char*)d_ws;
    auto carve = [&](size_t bytes) { void* r = (void*)p; p += alignUp(bytes); return r; };

    bf16_t* wbf_in  = (bf16_t*)carve((size_t)2 * D_INNER * D_MODEL * 2);   // 3072x768
    bf16_t* wbf_xp  = (bf16_t*)carve((size_t)XDBL * D_INNER * 2);          // 1664x1536
    bf16_t* wbf_dt  = (bf16_t*)carve((size_t)D_INNER * D_STATE * 2);       // 1536x64
    bf16_t* wbf_out = (bf16_t*)carve((size_t)D_MODEL * D_INNER * 2);       // 768x1536
    float*  x_cur   = (float*) carve((size_t)M_TOK * D_MODEL * 4);
    bf16_t* x_bf    = (bf16_t*)carve((size_t)M_TOK * D_MODEL * 2);
    float*  xr      = (float*) carve((size_t)M_TOK * 2 * D_INNER * 4);
    float*  xi_f    = (float*) carve((size_t)M_TOK * D_INNER * 4);
    bf16_t* xi_bf   = (bf16_t*)carve((size_t)M_TOK * D_INNER * 2);
    float*  x_dbl   = (float*) carve((size_t)M_TOK * XDBL * 4);
    bf16_t* dlt_bf  = (bf16_t*)carve((size_t)M_TOK * D_STATE * 2);
    float*  dt_raw  = (float*) carve((size_t)M_TOK * D_INNER * 4);
    float*  y_scan  = (float*) carve((size_t)M_TOK * D_INNER * 4);
    bf16_t* y_bf    = (bf16_t*)carve((size_t)M_TOK * D_INNER * 2);
    float*  outp    = (float*) carve((size_t)M_TOK * D_MODEL * 4);

    auto cvt = [&](const float* s, bf16_t* d, int n) {
        k_cvt_bf16<<<(n + 255) / 256, 256, 0, stream>>>(s, d, n);
    };
    auto gemm = [&](const bf16_t* A, const bf16_t* W, float* C, int M, int N, int K) {
        dim3 grid(N / 128, M / 32);     // 4 waves/block, 32x32 per wave
        k_gemm_wmma_bf16<<<grid, 128, 0, stream>>>(A, W, C, M, N, K);
    };

    // x_cur = x (never mutate inputs)
    hipMemcpyAsync(x_cur, x_in, (size_t)M_TOK * D_MODEL * 4,
                   hipMemcpyDeviceToDevice, stream);

    const int nInner = M_TOK * D_INNER;   // 1,572,864

    for (int i = 0; i < N_LAYERS; ++i) {
        // per-layer weight views -> bf16
        cvt(in_w  + (size_t)i * 2 * D_INNER * D_MODEL, wbf_in,  2 * D_INNER * D_MODEL);
        cvt(xp_w  + (size_t)i * XDBL * D_INNER,        wbf_xp,  XDBL * D_INNER);
        cvt(dt_w  + (size_t)i * D_INNER * D_STATE,     wbf_dt,  D_INNER * D_STATE);
        cvt(out_w + (size_t)i * D_MODEL * D_INNER,     wbf_out, D_MODEL * D_INNER);
        cvt(x_cur, x_bf, M_TOK * D_MODEL);

        // in_proj: [1024,768] x [3072,768]^T -> xr [1024,3072]
        gemm(x_bf, wbf_in, xr, M_TOK, 2 * D_INNER, D_MODEL);

        // xi = silu(xr[:, :1536])
        k_silu_split<<<(nInner + 255) / 256, 256, 0, stream>>>(xr, xi_f, xi_bf, nInner);

        // x_proj: [1024,1536] x [1664,1536]^T -> x_dbl
        gemm(xi_bf, wbf_xp, x_dbl, M_TOK, XDBL, D_INNER);

        // dt input -> bf16
        k_extract_dltin<<<(M_TOK * D_STATE + 255) / 256, 256, 0, stream>>>(
            x_dbl, dlt_bf, M_TOK * D_STATE);

        // dt_proj: [1024,64] x [1536,64]^T -> dt_raw
        gemm(dlt_bf, wbf_dt, dt_raw, M_TOK, D_INNER, D_STATE);

        // delta = min(softplus(dt_raw + b), 10)
        k_delta<<<(nInner + 255) / 256, 256, 0, stream>>>(
            dt_raw, dt_b + (size_t)i * D_INNER, nInner);

        // selective scan: 2*1536 waves, 8 waves/block
        k_scan<<<(BATCH * D_INNER) / 8, 256, 0, stream>>>(
            dt_raw, x_dbl, xi_f, A_log + (size_t)i * D_INNER * D_STATE, y_scan);

        // y = (y + xi*D) * silu(res)
        k_ypost<<<(nInner + 255) / 256, 256, 0, stream>>>(
            y_scan, xi_f, xr, D_par + (size_t)i * D_INNER, y_bf, nInner);

        // out_proj: [1024,1536] x [768,1536]^T -> outp
        gemm(y_bf, wbf_out, outp, M_TOK, D_MODEL, D_INNER);

        // x_cur = LN(outp + x_cur)
        k_add_ln<<<M_TOK, 256, 0, stream>>>(
            outp, x_cur, ln_g + (size_t)i * D_MODEL, ln_b + (size_t)i * D_MODEL, x_cur);
    }

    // final layernorm -> d_out
    k_add_ln<<<M_TOK, 256, 0, stream>>>(x_cur, nullptr, fln_g, fln_b, out);
}
